// DWAModel_46780783788085
// MI455X (gfx1250) — matrix operand order repaired and checked by reading, not compiled, not run on hardware
//
#include <hip/hip_runtime.h>
#include <hip/hip_bf16.h>
#include <cstddef>

// ---------------------------------------------------------------------------
// DWA model forward for MI455X (gfx1250, wave32).
// All big matmuls use V_WMMA_F32_16X16X32_BF16 (bf16 in, f32 accumulate).
// Storage stays fp32; bf16 conversion happens on the global->LDS staging path
// (float4 global loads -> packed bf16 LDS stores), with register
// double-buffering so global latency overlaps the WMMAs.
// ---------------------------------------------------------------------------

typedef __bf16 v16bf  __attribute__((ext_vector_type(16)));
typedef float  v8f    __attribute__((ext_vector_type(8)));
typedef float  f32x4  __attribute__((ext_vector_type(4)));
typedef __bf16 bf16x4 __attribute__((ext_vector_type(4)));
typedef __bf16 bf16x2 __attribute__((ext_vector_type(2)));

#define B_     4
#define T_     1024
#define DA_    512
#define DFF_   2048
#define NH_    8
#define DH_    64
#define VOC_   32000
#define NPOOL_ 512
#define DKEY_  128
#define TOPK_  8

static __device__ __forceinline__ bf16x4 cvt4(f32x4 v) {
  bf16x4 r;
  r[0] = (__bf16)v[0]; r[1] = (__bf16)v[1];
  r[2] = (__bf16)v[2]; r[3] = (__bf16)v[3];
  return r;
}

// ------------------------------ embed + posenc ------------------------------
__global__ void embed_kernel(const int* __restrict__ ids,
                             const float* __restrict__ embW,
                             float* __restrict__ x) {
  int row = blockIdx.x;            // b*T + t
  int t   = row & (T_ - 1);
  int id  = ids[row];
  for (int d = threadIdx.x; d < DA_; d += blockDim.x) {
    float pe;
    if (d < DA_ / 2) {
      float ang = (float)t * powf(10000.f, -(2.f * d) / (float)DA_);
      pe = sinf(ang);
    } else {
      int i = d - DA_ / 2;
      float ang = (float)t * powf(10000.f, -(2.f * i) / (float)DA_);
      pe = cosf(ang);
    }
    x[(size_t)row * DA_ + d] = embW[(size_t)id * DA_ + d] + pe;
  }
}

// ------------------------------ layernorm (D=512) ---------------------------
__global__ void ln_kernel(const float* __restrict__ x, float* __restrict__ y) {
  int row = blockIdx.x;
  const float* xr = x + (size_t)row * DA_;
  __shared__ float red[256];
  float s = 0.f;
  for (int d = threadIdx.x; d < DA_; d += 256) s += xr[d];
  red[threadIdx.x] = s; __syncthreads();
  for (int off = 128; off > 0; off >>= 1) {
    if (threadIdx.x < off) red[threadIdx.x] += red[threadIdx.x + off];
    __syncthreads();
  }
  float mean = red[0] * (1.f / DA_); __syncthreads();
  float v = 0.f;
  for (int d = threadIdx.x; d < DA_; d += 256) { float t = xr[d] - mean; v += t * t; }
  red[threadIdx.x] = v; __syncthreads();
  for (int off = 128; off > 0; off >>= 1) {
    if (threadIdx.x < off) red[threadIdx.x] += red[threadIdx.x + off];
    __syncthreads();
  }
  float rstd = rsqrtf(red[0] * (1.f / DA_) + 1e-5f);
  for (int d = threadIdx.x; d < DA_; d += 256)
    y[(size_t)row * DA_ + d] = (xr[d] - mean) * rstd;
}

// ----------------------- causal softmax, row len = 1024 ---------------------
__global__ void softmax_kernel(float* __restrict__ s) {
  int row  = blockIdx.x;           // (b*H + h)*T + q
  int qpos = row & (T_ - 1);
  float* sr = s + (size_t)row * T_;
  const float scale = 0.125f;      // 1/sqrt(64)
  __shared__ float red[256];
  float m = -3.4e38f;
  for (int k = threadIdx.x; k <= qpos; k += 256) m = fmaxf(m, sr[k] * scale);
  red[threadIdx.x] = m; __syncthreads();
  for (int off = 128; off > 0; off >>= 1) {
    if (threadIdx.x < off) red[threadIdx.x] = fmaxf(red[threadIdx.x], red[threadIdx.x + off]);
    __syncthreads();
  }
  m = red[0]; __syncthreads();
  float sum = 0.f;
  for (int k = threadIdx.x; k < T_; k += 256) {
    float e = (k <= qpos) ? __expf(sr[k] * scale - m) : 0.f;
    sr[k] = e; sum += e;
  }
  red[threadIdx.x] = sum; __syncthreads();
  for (int off = 128; off > 0; off >>= 1) {
    if (threadIdx.x < off) red[threadIdx.x] += red[threadIdx.x + off];
    __syncthreads();
  }
  float inv = 1.f / red[0];
  for (int k = threadIdx.x; k < T_; k += 256) sr[k] *= inv;
}

// ------------------------------ gelu (tanh approx) --------------------------
__global__ void gelu_kernel(float* __restrict__ x) {
  size_t i = (size_t)blockIdx.x * 256 + threadIdx.x;
  float v = x[i];
  float c = 0.7978845608028654f * (v + 0.044715f * v * v * v);
  x[i] = 0.5f * v * (1.f + tanhf(c));
}

// ------------------------------ mean pool over T ----------------------------
__global__ void meanpool_kernel(const float* __restrict__ h, float* __restrict__ z) {
  int i = blockIdx.x * blockDim.x + threadIdx.x;  // 0..B*DA-1
  int b = i >> 9, d = i & (DA_ - 1);
  const float* p = h + (size_t)b * T_ * DA_ + d;
  float s = 0.f;
  for (int t = 0; t < T_; ++t) s += p[(size_t)t * DA_];
  z[i] = s * (1.f / T_);
}

// --------------------- retrieval: q proj, cosine, top-k ---------------------
__global__ void retrieval_kernel(const float* __restrict__ z,
                                 const float* __restrict__ Wq,
                                 const float* __restrict__ keys,
                                 const int* __restrict__ lam,
                                 const int* __restrict__ warm,
                                 float* __restrict__ meta,       // alphas f32[32] then idx i32[32]
                                 float* __restrict__ out_alphas,
                                 int* __restrict__ out_idx) {
  int b = blockIdx.x;
  int j = threadIdx.x;             // 128 threads == DKEY
  __shared__ float qsh[DKEY_];
  __shared__ float sc[NPOOL_];
  __shared__ float red[DKEY_];
  const float* zb = z + (size_t)b * DA_;
  float q = 0.f;
  for (int d = 0; d < DA_; ++d) q += zb[d] * Wq[(size_t)d * DKEY_ + j];
  red[j] = q * q; __syncthreads();
  for (int off = 64; off > 0; off >>= 1) {
    if (j < off) red[j] += red[j + off];
    __syncthreads();
  }
  float qn = q / (sqrtf(red[0]) + 1e-6f);
  qsh[j] = qn; __syncthreads();
  float lambda = (float)lam[0];
  for (int n = j; n < NPOOL_; n += DKEY_) {
    const float* kn = keys + (size_t)n * DKEY_;
    float dot = 0.f, nn = 0.f;
    for (int d = 0; d < DKEY_; ++d) { float kv = kn[d]; dot += qsh[d] * kv; nn += kv * kv; }
    sc[n] = lambda * dot / (sqrtf(nn) + 1e-6f);
  }
  __syncthreads();
  if (j == 0) {
    float vals[TOPK_]; int idx[TOPK_];
    for (int k = 0; k < TOPK_; ++k) {
      float best = -3.4e38f; int bi = 0;
      for (int n = 0; n < NPOOL_; ++n) if (sc[n] > best) { best = sc[n]; bi = n; }
      vals[k] = best; idx[k] = bi; sc[bi] = -3.4e38f;
    }
    float alphas[TOPK_];
    if (warm[0]) {
      for (int k = 0; k < TOPK_; ++k) alphas[k] = 1.f / TOPK_;
    } else {
      float mx = vals[0], s = 0.f;
      for (int k = 0; k < TOPK_; ++k) { alphas[k] = __expf(vals[k] - mx); s += alphas[k]; }
      for (int k = 0; k < TOPK_; ++k) alphas[k] /= s;
    }
    for (int k = 0; k < TOPK_; ++k) {
      meta[b * TOPK_ + k] = alphas[k];
      ((int*)meta)[32 + b * TOPK_ + k] = idx[k];
      out_alphas[b * TOPK_ + k] = alphas[k];
      out_idx[b * TOPK_ + k] = idx[k];
    }
  }
}

// ------------------ W[b] = W_base + sum_k alpha * pool[idx] -----------------
__global__ void assembleW_kernel(const float* __restrict__ Wbase,
                                 const float* __restrict__ pool,
                                 const float* __restrict__ meta,
                                 float* __restrict__ Wb) {
  size_t i = (size_t)blockIdx.x * 256 + threadIdx.x;  // 0..B*DA*DA-1
  int b = (int)(i >> 18);
  size_t e = i & ((1u << 18) - 1);
  const int* idx = ((const int*)meta) + 32;
  float acc = Wbase[e];
  #pragma unroll
  for (int k = 0; k < TOPK_; ++k)
    acc += meta[b * TOPK_ + k] * pool[(size_t)idx[b * TOPK_ + k] * (DA_ * DA_) + e];
  Wb[i] = acc;
}

// ---------------------------------------------------------------------------
// Batched GEMM: C = (acc ? C : 0) + A * B, bf16 WMMA, f32 accumulate.
// BMxBN workgroup tile, 256 threads (8 waves), BK=32.
// Wave w: msubs (w%4)*MS..+MS-1, nsubs (w/4)*NS..+NS-1  (MS=BM/64, NS=BN/32).
// Requires M%BM==0, N%BN==0, K%32==0, and lda/ldb multiples of 4 (true here).
// TRANSB: B stored [N,K]; else [K,N].
//   transB=0 staging: each thread owns a BV(k) x 4(n) micro-block so each of
//   its 4 N-columns is K-contiguous in LDS -> ds_store_b64/b32, no scatter.
// Dual batch index: z -> (b1=z/nb2, b2=z%nb2), offsets sX1*b1 + sX2*b2.
// ---------------------------------------------------------------------------
template <int BM, int BN, int TRANSB>
__global__ __launch_bounds__(256)
void gemm_wmma_kernel(const float* __restrict__ A, const float* __restrict__ B,
                      float* __restrict__ C,
                      int K, int lda, int ldb, int ldc,
                      long sA1, long sA2, long sB1, long sB2, long sC1, long sC2,
                      int nb2, int acc) {
  constexpr int AV = BM / 32;          // float4 loads/thread for the A tile
  constexpr int BV = BN / 32;          // float4 loads/thread for the B tile
  constexpr int MS = BM / 64;          // 16x16 m-subtiles per wave
  constexpr int NS = BN / 32;          // 16x16 n-subtiles per wave

  const int bz = blockIdx.z;
  const int b1 = bz / nb2, b2 = bz % nb2;
  A += (size_t)b1 * sA1 + (size_t)b2 * sA2;
  B += (size_t)b1 * sB1 + (size_t)b2 * sB2;
  C += (size_t)b1 * sC1 + (size_t)b2 * sC2;

  const int m0 = blockIdx.y * BM;
  const int n0 = blockIdx.x * BN;

  // stride 48 elems (96B): 16B-aligned rows, no pathological bank conflicts
  __shared__ __bf16 As[BM][48];        // [m][k]
  __shared__ __bf16 Bs[BN][48];        // [n][k]  (K-contiguous per row)

  const int tid  = threadIdx.x;
  const int lane = tid & 31;
  const int wave = tid >> 5;
  const int msb  = (wave & 3) * MS;    // first m-subtile of this wave
  const int nsb  = (wave >> 2) * NS;   // first n-subtile of this wave

  const int hi8  = (lane >= 16) ? 8 : 0;   // C/D: lanes 16-31 hold M+8
  const int ncol = lane & 15;
  const int ah   = (lane >= 16) ? 8 : 0;   // unused: A frag also uses kh below
  const int kh   = (lane >= 16) ? 16 : 0;  // B frag K offset for lanes 16-31

  // ---- A staging addresses: contiguous along K, bf16x4 stores ----
  int aRow[AV], aCol[AV];
  const float* aPtr[AV];
  #pragma unroll
  for (int i = 0; i < AV; ++i) {
    int idx = tid + i * 256;             // over BM*8 float4 slots
    aRow[i] = idx >> 3;
    aCol[i] = (idx & 7) * 4;
    aPtr[i] = A + (size_t)(m0 + aRow[i]) * lda + aCol[i];
  }
  // ---- B staging addresses ----
  int bRow[BV], bCol[BV];
  const float* bPtr[BV];
  int bK = 0, bN = 0;                    // micro-block base (transB==0 path)
  const float* bBlk = nullptr;
  if (TRANSB) {                          // B[N,K]: contiguous along K
    #pragma unroll
    for (int i = 0; i < BV; ++i) {
      int idx = tid + i * 256;
      bRow[i] = idx >> 3;                // n within tile
      bCol[i] = (idx & 7) * 4;           // k within tile
      bPtr[i] = B + (size_t)(n0 + bRow[i]) * ldb + bCol[i];
    }
  } else {                               // B[K,N]: BVx4 micro-block per thread
    bK = (tid / (BN / 4)) * BV;          // k base (0..32-BV)
    bN = (tid % (BN / 4)) * 4;           // n base
    bBlk = B + (size_t)bK * ldb + n0 + bN;
  }

  // ---- accumulators ----
  v8f accv[MS][NS];
  #pragma unroll
  for (int ms = 0; ms < MS; ++ms)
    #pragma unroll
    for (int ns = 0; ns < NS; ++ns) {
      if (acc) {
        #pragma unroll
        for (int r = 0; r < 8; ++r)
          accv[ms][ns][r] =
              C[(size_t)(m0 + (msb + ms) * 16 + r + hi8) * ldc +
                n0 + (nsb + ns) * 16 + ncol];
      } else {
        accv[ms][ns] = (v8f){0.f, 0.f, 0.f, 0.f, 0.f, 0.f, 0.f, 0.f};
      }
    }

  // ---- prologue: load first tiles into registers ----
  f32x4 ar[AV], br[BV];
  #pragma unroll
  for (int i = 0; i < AV; ++i) ar[i] = *(const f32x4*)aPtr[i];
  if (TRANSB) {
    #pragma unroll
    for (int i = 0; i < BV; ++i) br[i] = *(const f32x4*)bPtr[i];
  } else {
    #pragma unroll
    for (int i = 0; i < BV; ++i) br[i] = *(const f32x4*)(bBlk + (size_t)i * ldb);
  }

  for (int k0 = 0; k0 < K; k0 += 32) {
    __syncthreads();                     // previous compute done reading LDS
    // ---- convert + store staged registers to LDS ----
    #pragma unroll
    for (int i = 0; i < AV; ++i)
      *(bf16x4*)&As[aRow[i]][aCol[i]] = cvt4(ar[i]);
    if (TRANSB) {
      #pragma unroll
      for (int i = 0; i < BV; ++i)
        *(bf16x4*)&Bs[bRow[i]][bCol[i]] = cvt4(br[i]);
    } else {
      // register transpose of the BVx4 block: column j is K-contiguous
      #pragma unroll
      for (int j = 0; j < 4; ++j) {
        if (BV == 4) {
          bf16x4 v;
          v[0] = (__bf16)br[0][j]; v[1] = (__bf16)br[1][j];
          v[2] = (__bf16)br[2][j]; v[3] = (__bf16)br[3][j];
          *(bf16x4*)&Bs[bN + j][bK] = v;
        } else {
          bf16x2 v;
          v[0] = (__bf16)br[0][j]; v[1] = (__bf16)br[1][j];
          *(bf16x2*)&Bs[bN + j][bK] = v;
        }
      }
    }
    __syncthreads();

    // ---- prefetch next K-step while WMMAs run ----
    if (k0 + 32 < K) {
      #pragma unroll
      for (int i = 0; i < AV; ++i) { aPtr[i] += 32; ar[i] = *(const f32x4*)aPtr[i]; }
      if (TRANSB) {
        #pragma unroll
        for (int i = 0; i < BV; ++i) { bPtr[i] += 32; br[i] = *(const f32x4*)bPtr[i]; }
      } else {
        bBlk += (size_t)32 * ldb;
        #pragma unroll
        for (int i = 0; i < BV; ++i) br[i] = *(const f32x4*)(bBlk + (size_t)i * ldb);
      }
    }

    // ---- fragments + WMMA ----
    v16bf bfrag[NS];
    #pragma unroll
    for (int ns = 0; ns < NS; ++ns) {
      int bc = ncol + (nsb + ns) * 16;
      #pragma unroll
      for (int e = 0; e < 16; ++e) bfrag[ns][e] = Bs[bc][kh + e];
    }
    #pragma unroll
    for (int ms = 0; ms < MS; ++ms) {
      v16bf afrag;
      int arw = ncol + (msb + ms) * 16;
      #pragma unroll
      for (int e = 0; e < 16; ++e) {
        int kb = kh + e;
        afrag[e] = As[arw][kb];
      }
      #pragma unroll
      for (int ns = 0; ns < NS; ++ns)
        accv[ms][ns] = __builtin_amdgcn_wmma_f32_16x16x32_bf16(
            false, afrag, false, bfrag[ns], (short)0, accv[ms][ns], false, false);
    }
  }

  // ---- store C ----
  #pragma unroll
  for (int ms = 0; ms < MS; ++ms)
    #pragma unroll
    for (int ns = 0; ns < NS; ++ns)
      #pragma unroll
      for (int r = 0; r < 8; ++r)
        C[(size_t)(m0 + (msb + ms) * 16 + r + hi8) * ldc +
          n0 + (nsb + ns) * 16 + ncol] = accv[ms][ns][r];
}

// ------------------------------ host-side helpers ---------------------------
template <int BM, int BN, int TRANSB>
static inline void gemm_t(hipStream_t st, const float* A, const float* B, float* C,
                          int M, int N, int K, int lda, int ldb, int ldc,
                          long sA1, long sA2, long sB1, long sB2, long sC1, long sC2,
                          int nb, int nb2, int acc) {
  dim3 g(N / BN, M / BM, nb * nb2);
  gemm_wmma_kernel<BM, BN, TRANSB><<<g, dim3(256), 0, st>>>(
      A, B, C, K, lda, ldb, ldc, sA1, sA2, sB1, sB2, sC1, sC2, nb2, acc);
}

static void run_block(hipStream_t st, float* h,
                      const float* Wqkv, const float* Wo,
                      const float* Wf1, const float* Wf2,
                      float* ln, float* qkv, float* scores, float* o, float* ff1) {
  const int MT = B_ * T_;          // 4096 token rows
  // pre-LN -> QKV
  ln_kernel<<<MT, 256, 0, st>>>(h, ln);
  gemm_t<128, 128, 0>(st, ln, Wqkv, qkv, MT, 3 * DA_, DA_, DA_, 3 * DA_, 3 * DA_,
                      0, 0, 0, 0, 0, 0, 1, 1, 0);
  // scores[b,h] = q @ k^T   (batch z -> b1=batch, b2=head), B stored [N,K]
  gemm_t<128, 128, 1>(st, qkv, qkv + DA_, scores, T_, T_, DH_, 3 * DA_, 3 * DA_, T_,
                      (long)T_ * 3 * DA_, DH_, (long)T_ * 3 * DA_, DH_,
                      (long)NH_ * T_ * T_, (long)T_ * T_, B_, NH_, 0);
  softmax_kernel<<<B_ * NH_ * T_, 256, 0, st>>>(scores);
  // o[b,h] = attn @ v   (N = 64 -> 64x64 tile config)
  gemm_t<64, 64, 0>(st, scores, qkv + 2 * DA_, o, T_, DH_, T_, T_, 3 * DA_, DA_,
                    (long)NH_ * T_ * T_, (long)T_ * T_, (long)T_ * 3 * DA_, DH_,
                    (long)T_ * DA_, DH_, B_, NH_, 0);
  // h += o @ Wo   (fused residual via acc)
  gemm_t<128, 128, 0>(st, o, Wo, h, MT, DA_, DA_, DA_, DA_, DA_,
                      0, 0, 0, 0, 0, 0, 1, 1, 1);
  // FFN: h += gelu(LN(h) @ Wf1) @ Wf2
  ln_kernel<<<MT, 256, 0, st>>>(h, ln);
  gemm_t<128, 128, 0>(st, ln, Wf1, ff1, MT, DFF_, DA_, DA_, DFF_, DFF_,
                      0, 0, 0, 0, 0, 0, 1, 1, 0);
  gelu_kernel<<<(MT * DFF_) / 256, 256, 0, st>>>(ff1);
  gemm_t<128, 128, 0>(st, ff1, Wf2, h, MT, DA_, DFF_, DFF_, DA_, DA_,
                      0, 0, 0, 0, 0, 0, 1, 1, 1);
}

// ------------------------------ entry point ---------------------------------
extern "C" void kernel_launch(void* const* d_in, const int* in_sizes, int n_in,
                              void* d_out, int out_size, void* d_ws, size_t ws_size,
                              hipStream_t stream) {
  const int*   ids     = (const int*)  d_in[0];
  const int*   lam     = (const int*)  d_in[1];
  const int*   warm    = (const int*)  d_in[2];
  const float* embW    = (const float*)d_in[3];
  const float* Wqkv_a  = (const float*)d_in[4];
  const float* Wo_a    = (const float*)d_in[5];
  const float* Wf1_a   = (const float*)d_in[6];
  const float* Wf2_a   = (const float*)d_in[7];
  const float* pool_v  = (const float*)d_in[8];
  const float* pool_k  = (const float*)d_in[9];
  const float* Wq_ret  = (const float*)d_in[10];
  const float* W_base  = (const float*)d_in[11];
  const float* Wqkv_b  = (const float*)d_in[12];
  const float* Wo_b    = (const float*)d_in[13];
  const float* Wf1_b   = (const float*)d_in[14];
  const float* Wf2_b   = (const float*)d_in[15];
  const float* lmW     = (const float*)d_in[16];

  // Workspace layout (f32 elements)
  float* ws = (float*)d_ws;
  const size_t MT = (size_t)B_ * T_;
  float* H      = ws;                          size_t off = MT * DA_;
  float* LN     = ws + off;                    off += MT * DA_;
  float* QKV    = ws + off;                    off += MT * 3 * DA_;
  float* SCORES = ws + off;                    off += (size_t)B_ * NH_ * T_ * T_;
  float* O      = ws + off;                    off += MT * DA_;
  float* FF1    = ws + off;                    off += MT * DFF_;
  float* Z      = ws + off;                    off += B_ * DA_;
  float* META   = ws + off;                    off += 64;
  float* WB     = ws + off;                    off += (size_t)B_ * DA_ * DA_;
  float* HMID   = ws + off;                    off += MT * DA_;
  (void)off; (void)ws_size; (void)n_in; (void)in_sizes; (void)out_size;

  // Output layout: logits f32[B*T*VOC], alphas f32[B*K], indices i32[B*K]
  float* logits     = (float*)d_out;
  float* out_alphas = logits + (size_t)B_ * T_ * VOC_;
  int*   out_idx    = (int*)(out_alphas + B_ * TOPK_);

  // 1) embed + posenc
  embed_kernel<<<(int)MT, 256, 0, stream>>>(ids, embW, H);
  // 2) Part A transformer block
  run_block(stream, H, Wqkv_a, Wo_a, Wf1_a, Wf2_a, LN, QKV, SCORES, O, FF1);
  // 3) mean pool + retrieval + per-example weight assembly
  meanpool_kernel<<<(B_ * DA_) / 256, 256, 0, stream>>>(H, Z);
  retrieval_kernel<<<B_, DKEY_, 0, stream>>>(Z, Wq_ret, pool_k, lam, warm,
                                             META, out_alphas, out_idx);
  assembleW_kernel<<<(B_ * DA_ * DA_) / 256, 256, 0, stream>>>(W_base, pool_v, META, WB);
  // 4) h_mid[b] = h_A[b] @ W[b]   (batched over B)
  gemm_t<128, 128, 0>(stream, H, WB, HMID, T_, DA_, DA_, DA_, DA_, DA_,
                      (long)T_ * DA_, 0, (long)DA_ * DA_, 0, (long)T_ * DA_, 0,
                      B_, 1, 0);
  // 5) Part B transformer block
  run_block(stream, HMID, Wqkv_b, Wo_b, Wf1_b, Wf2_b, LN, QKV, SCORES, O, FF1);
  // 6) LM head: logits = h_out @ lm_head_W  (134 GF, the dominant GEMM)
  gemm_t<128, 128, 0>(stream, HMID, lmW, logits, (int)MT, VOC_, DA_, DA_, VOC_, VOC_,
                      0, 0, 0, 0, 0, 0, 1, 1, 0);
}
